// FlatConv3x3NNUE_46892452938213
// MI455X (gfx1250) — matrix operand order repaired and checked by reading, not compiled, not run on hardware
//
#include <hip/hip_runtime.h>
#include <hip/hip_bf16.h>

typedef _Float16 h8  __attribute__((ext_vector_type(8)));
typedef _Float16 h16 __attribute__((ext_vector_type(16)));
typedef float    f8  __attribute__((ext_vector_type(8)));

namespace {
constexpr int    P_    = 25;
constexpr int    B_    = 8192;
constexpr int    DF_   = 32;
constexpr float  CMAX_ = 127.0f / 128.0f;
constexpr int    FRAG  = 512;   // f16 elements per packed B-fragment (32 lanes x 16)

// ---- workspace layout (bytes) ----
constexpr size_t FEAT_OFF   = 0;                               // 5 groups * B * DF f32
constexpr size_t FEAT_BYTES = 5ull * B_ * DF_ * 4;
constexpr size_t W1E_N      = 25ull * 4 * 8 * FRAG;            // [p][spatial][nt] frags
constexpr size_t W1E_OFF    = FEAT_OFF + FEAT_BYTES;
constexpr size_t W2E_N      = 25ull * 16 * 8 * FRAG;           // [p][kstep][nt]
constexpr size_t W2E_OFF    = W1E_OFF + W1E_N * 2;
constexpr size_t W3E_N      = 25ull * 4 * 8 * FRAG;            // [p][kstep][nt]
constexpr size_t W3E_OFF    = W2E_OFF + W2E_N * 2;
constexpr size_t W4E_N      = 25ull * 4 * 2 * FRAG;            // [p][kstep][nt]
constexpr size_t W4E_OFF    = W3E_OFF + W3E_N * 2;

constexpr int ROWS = 128;     // batch rows per block (4 waves x 2 M-tiles x 16)
} // namespace

// =====================================================================
// Weight repack: f32 reference layouts -> f16 WMMA-B fragment layout.
// Fragment element (lane, i): B(K = kbase + ((lane>=16)?16:0) + i,
//                               N = ntile*16 + (lane&15))
// =====================================================================
__global__ void prep_w1(const float* __restrict__ W1, _Float16* __restrict__ w1e) {
  size_t idx = (size_t)blockIdx.x * blockDim.x + threadIdx.x;
  if (idx >= W1E_N) return;
  int i    = (int)(idx & 15);
  int lane = (int)((idx >> 4) & 31);
  int frag = (int)(idx >> 9);
  int nt = frag & 7;
  int s  = (frag >> 3) & 3;          // output spatial = si*2+sj
  int p  = frag >> 5;
  int K  = ((lane >> 4) << 4) + i;   // 0..31 (18 real)
  int o  = nt * 16 + (lane & 15);
  float v = 0.f;
  if (K < 18) {
    int c = K / 9, q = K % 9;
    int y = q / 3, xx = q % 3;
    int si = s >> 1, sj = s & 1;
    int a = y - si, b = xx - sj;
    if (a >= 0 && a < 2 && b >= 0 && b < 2)
      v = W1[(((size_t)(p * 128 + o) * 2 + c) * 2 + a) * 2 + b];
  }
  w1e[idx] = (_Float16)v;
}

__global__ void prep_w2(const float* __restrict__ W2, _Float16* __restrict__ w2e) {
  size_t idx = (size_t)blockIdx.x * blockDim.x + threadIdx.x;
  if (idx >= W2E_N) return;
  int i    = (int)(idx & 15);
  int lane = (int)((idx >> 4) & 31);
  int frag = (int)(idx >> 9);
  int nt = frag & 7;
  int ks = (frag >> 3) & 15;
  int p  = frag >> 7;
  int K  = ks * 32 + ((lane >> 4) << 4) + i;   // 0..511: spatial*128 + c
  int s  = K >> 7;
  int c  = K & 127;
  int o  = nt * 16 + (lane & 15);
  w2e[idx] = (_Float16)W2[((size_t)(p * 128 + o) * 128 + c) * 4 + s];
}

__global__ void prep_w3(const float* __restrict__ W3, _Float16* __restrict__ w3e) {
  size_t idx = (size_t)blockIdx.x * blockDim.x + threadIdx.x;
  if (idx >= W3E_N) return;
  int i    = (int)(idx & 15);
  int lane = (int)((idx >> 4) & 31);
  int frag = (int)(idx >> 9);
  int nt = frag & 7;
  int ks = (frag >> 3) & 3;
  int p  = frag >> 5;
  int c  = ks * 32 + ((lane >> 4) << 4) + i;
  int o  = nt * 16 + (lane & 15);
  w3e[idx] = (_Float16)W3[(size_t)(p * 128 + o) * 128 + c];
}

__global__ void prep_w4(const float* __restrict__ W4, _Float16* __restrict__ w4e) {
  size_t idx = (size_t)blockIdx.x * blockDim.x + threadIdx.x;
  if (idx >= W4E_N) return;
  int i    = (int)(idx & 15);
  int lane = (int)((idx >> 4) & 31);
  int frag = (int)(idx >> 9);
  int nt = frag & 1;
  int ks = (frag >> 1) & 3;
  int p  = frag >> 3;
  int c  = ks * 32 + ((lane >> 4) << 4) + i;
  int o  = nt * 16 + (lane & 15);
  w4e[idx] = (_Float16)W4[(size_t)(p * 32 + o) * 128 + c];
}

// =====================================================================
// Tower kernel: WMMA towers, 2 M-tiles per wave for 2x B-fragment reuse.
// =====================================================================
__device__ inline h16 load_a16(const _Float16* base) {
  // A order used here (same as B packing): lane's 16 halves are 16 contiguous K at
  // +0..+15 (K offset kb = 16*half is folded into `base` by caller).
  h8 lo = *(const h8*)(base);
  h8 hv = *(const h8*)(base + 8);
  return __builtin_shufflevector(lo, hv, 0,1,2,3,4,5,6,7,8,9,10,11,12,13,14,15);
}

__global__ __launch_bounds__(128) void tower_kernel(
    const float* __restrict__ x,
    const float* __restrict__ b1v, const float* __restrict__ b2v,
    const float* __restrict__ b3v, const float* __restrict__ b4v,
    const _Float16* __restrict__ w1e, const _Float16* __restrict__ w2e,
    const _Float16* __restrict__ w3e, const _Float16* __restrict__ w4e,
    float* __restrict__ featpart)
{
  extern __shared__ _Float16 smem[];
  _Float16* patch = smem;               // [ROWS][32]  f16
  _Float16* hbuf  = smem + ROWS * 32;   // [ROWS][512] f16

  const int tid   = threadIdx.x;
  const int wave  = tid >> 5;
  const int lane  = tid & 31;
  const int nloc  = lane & 15;
  const int hi    = lane >> 4;            // K-half selector
  const int kb    = hi << 4;              // A chunk base: 0 or 16
  const int rowl0 = wave * 32 + nloc;     // A row, M-tile 0
  const int rowl1 = rowl0 + 16;           // A row, M-tile 1
  const int wst0  = wave * 32 + hi * 8;   // C store row base, M-tile 0
  const int rows0 = blockIdx.x * ROWS;
  const int grp   = blockIdx.y;           // position group (5 positions each)

  const f8 fzero = {};
  f8 facc[2][2];
  facc[0][0] = fzero; facc[0][1] = fzero;
  facc[1][0] = fzero; facc[1][1] = fzero;

  for (int pi = 0; pi < 5; ++pi) {
    const int p  = grp * 5 + pi;
    const int py = p / 5, px = p % 5;

    __syncthreads();
    // Build 3x3x2 patches (18 real K, padded to 32) for ROWS batch rows.
    for (int idx = tid; idx < ROWS * 32; idx += 128) {
      int r = idx >> 5, k = idx & 31;
      float v = 0.f;
      if (k < 18) {
        int c = k / 9, q = k % 9;
        v = x[((size_t)(rows0 + r) * 2 + c) * 49 + (py + q / 3) * 7 + (px + q % 3)];
      }
      patch[idx] = (_Float16)v;
    }
    __syncthreads();

    // ---- conv1: 4 spatial GEMMs, M=32 N=128 K=32(18) -> hbuf[row][s*128+c]
    const h16 a1t0 = load_a16(patch + rowl0 * 32 + kb);
    const h16 a1t1 = load_a16(patch + rowl1 * 32 + kb);
    #pragma unroll
    for (int s = 0; s < 4; ++s) {
      const _Float16* wb = w1e + ((size_t)((p * 4 + s) * 8) * 32 + lane) * 16;
      #pragma unroll
      for (int nt = 0; nt < 8; ++nt) {
        h16 bf = *(const h16*)(wb + (size_t)nt * FRAG);
        f8 c0 = {}, c1 = {};
        c0 = __builtin_amdgcn_wmma_f32_16x16x32_f16(false, a1t0, false, bf,
                                                    (short)0, c0, false, false);
        c1 = __builtin_amdgcn_wmma_f32_16x16x32_f16(false, a1t1, false, bf,
                                                    (short)0, c1, false, false);
        float bias = b1v[p * 128 + nt * 16 + nloc];
        #pragma unroll
        for (int r = 0; r < 8; ++r) {
          float v0 = c0[r] + bias; v0 = v0 > 0.f ? v0 : 0.f;
          float v1 = c1[r] + bias; v1 = v1 > 0.f ? v1 : 0.f;
          hbuf[(size_t)(wst0 + r) * 512 + s * 128 + nt * 16 + nloc]      = (_Float16)v0;
          hbuf[(size_t)(wst0 + 16 + r) * 512 + s * 128 + nt * 16 + nloc] = (_Float16)v1;
        }
      }
    }

    // ---- conv2: M=32 N=128 K=512 -> hbuf[row][0..127]
    f8 acc[2][8];
    #pragma unroll
    for (int nt = 0; nt < 8; ++nt) { acc[0][nt] = fzero; acc[1][nt] = fzero; }
    for (int ks = 0; ks < 16; ++ks) {
      h16 a0 = load_a16(hbuf + rowl0 * 512 + ks * 32 + kb);
      h16 a1 = load_a16(hbuf + rowl1 * 512 + ks * 32 + kb);
      const _Float16* wb = w2e + ((size_t)((p * 16 + ks) * 8) * 32 + lane) * 16;
      #pragma unroll
      for (int nt = 0; nt < 8; ++nt) {
        h16 bf = *(const h16*)(wb + (size_t)nt * FRAG);
        acc[0][nt] = __builtin_amdgcn_wmma_f32_16x16x32_f16(
            false, a0, false, bf, (short)0, acc[0][nt], false, false);
        acc[1][nt] = __builtin_amdgcn_wmma_f32_16x16x32_f16(
            false, a1, false, bf, (short)0, acc[1][nt], false, false);
      }
    }
    #pragma unroll
    for (int nt = 0; nt < 8; ++nt) {
      float bias = b2v[p * 128 + nt * 16 + nloc];
      #pragma unroll
      for (int r = 0; r < 8; ++r) {
        float v0 = acc[0][nt][r] + bias; v0 = v0 > 0.f ? v0 : 0.f;
        float v1 = acc[1][nt][r] + bias; v1 = v1 > 0.f ? v1 : 0.f;
        hbuf[(size_t)(wst0 + r) * 512 + nt * 16 + nloc]      = (_Float16)v0;
        hbuf[(size_t)(wst0 + 16 + r) * 512 + nt * 16 + nloc] = (_Float16)v1;
      }
    }

    // ---- conv3: M=32 N=128 K=128 -> hbuf[row][128..255]
    #pragma unroll
    for (int nt = 0; nt < 8; ++nt) { acc[0][nt] = fzero; acc[1][nt] = fzero; }
    #pragma unroll
    for (int ks = 0; ks < 4; ++ks) {
      h16 a0 = load_a16(hbuf + rowl0 * 512 + ks * 32 + kb);
      h16 a1 = load_a16(hbuf + rowl1 * 512 + ks * 32 + kb);
      const _Float16* wb = w3e + ((size_t)((p * 4 + ks) * 8) * 32 + lane) * 16;
      #pragma unroll
      for (int nt = 0; nt < 8; ++nt) {
        h16 bf = *(const h16*)(wb + (size_t)nt * FRAG);
        acc[0][nt] = __builtin_amdgcn_wmma_f32_16x16x32_f16(
            false, a0, false, bf, (short)0, acc[0][nt], false, false);
        acc[1][nt] = __builtin_amdgcn_wmma_f32_16x16x32_f16(
            false, a1, false, bf, (short)0, acc[1][nt], false, false);
      }
    }
    #pragma unroll
    for (int nt = 0; nt < 8; ++nt) {
      float bias = b3v[p * 128 + nt * 16 + nloc];
      #pragma unroll
      for (int r = 0; r < 8; ++r) {
        float v0 = acc[0][nt][r] + bias; v0 = v0 > 0.f ? v0 : 0.f;
        float v1 = acc[1][nt][r] + bias; v1 = v1 > 0.f ? v1 : 0.f;
        hbuf[(size_t)(wst0 + r) * 512 + 128 + nt * 16 + nloc]      = (_Float16)v0;
        hbuf[(size_t)(wst0 + 16 + r) * 512 + 128 + nt * 16 + nloc] = (_Float16)v1;
      }
    }

    // ---- conv4: M=32 N=32 K=128, clamp [-1, 127/128], accumulate feature
    f8 acc4[2][2];
    acc4[0][0] = fzero; acc4[0][1] = fzero;
    acc4[1][0] = fzero; acc4[1][1] = fzero;
    #pragma unroll
    for (int ks = 0; ks < 4; ++ks) {
      h16 a0 = load_a16(hbuf + rowl0 * 512 + 128 + ks * 32 + kb);
      h16 a1 = load_a16(hbuf + rowl1 * 512 + 128 + ks * 32 + kb);
      const _Float16* wb = w4e + ((size_t)((p * 4 + ks) * 2) * 32 + lane) * 16;
      #pragma unroll
      for (int nt = 0; nt < 2; ++nt) {
        h16 bf = *(const h16*)(wb + (size_t)nt * FRAG);
        acc4[0][nt] = __builtin_amdgcn_wmma_f32_16x16x32_f16(
            false, a0, false, bf, (short)0, acc4[0][nt], false, false);
        acc4[1][nt] = __builtin_amdgcn_wmma_f32_16x16x32_f16(
            false, a1, false, bf, (short)0, acc4[1][nt], false, false);
      }
    }
    #pragma unroll
    for (int t = 0; t < 2; ++t) {
      #pragma unroll
      for (int nt = 0; nt < 2; ++nt) {
        float bias = b4v[p * 32 + nt * 16 + nloc];
        #pragma unroll
        for (int r = 0; r < 8; ++r) {
          float v = acc4[t][nt][r] + bias;
          v = v < -1.f ? -1.f : (v > CMAX_ ? CMAX_ : v);
          facc[t][nt][r] += v;
        }
      }
    }
  }

  // One writer per element -> no atomics, deterministic.
  #pragma unroll
  for (int t = 0; t < 2; ++t) {
    #pragma unroll
    for (int nt = 0; nt < 2; ++nt) {
      #pragma unroll
      for (int r = 0; r < 8; ++r) {
        int grow = rows0 + wst0 + t * 16 + r;
        featpart[((size_t)grp * B_ + grow) * DF_ + nt * 16 + nloc] = facc[t][nt][r];
      }
    }
  }
}

// =====================================================================
// Head: sum 5 partials, clipped-ReLU MLP 32->32->32->1.
// =====================================================================
__global__ __launch_bounds__(256) void head_kernel(
    const float* __restrict__ featpart,
    const float* __restrict__ fc1w, const float* __restrict__ fc1b,
    const float* __restrict__ fc2w, const float* __restrict__ fc2b,
    const float* __restrict__ fc3w, const float* __restrict__ fc3b,
    float* __restrict__ out)
{
  __shared__ float w1s[DF_ * DF_];
  __shared__ float w2s[DF_ * DF_];
  __shared__ float w3s[DF_], b1s[DF_], b2s[DF_];
  __shared__ float b3s;
  const int tid = threadIdx.x;
  for (int i = tid; i < DF_ * DF_; i += 256) { w1s[i] = fc1w[i]; w2s[i] = fc2w[i]; }
  if (tid < DF_) { w3s[tid] = fc3w[tid]; b1s[tid] = fc1b[tid]; b2s[tid] = fc2b[tid]; }
  if (tid == 0) b3s = fc3b[0];
  __syncthreads();

  const int row = blockIdx.x * 256 + tid;
  float v[DF_], u[DF_];
  #pragma unroll
  for (int j = 0; j < DF_; ++j) {
    float s = 0.f;
    #pragma unroll
    for (int g = 0; g < 5; ++g) s += featpart[((size_t)g * B_ + row) * DF_ + j];
    v[j] = s < 0.f ? 0.f : (s > CMAX_ ? CMAX_ : s);
  }
  for (int i = 0; i < DF_; ++i) {
    float s = b1s[i];
    #pragma unroll
    for (int j = 0; j < DF_; ++j) s += v[j] * w1s[i * DF_ + j];
    u[i] = s < 0.f ? 0.f : (s > CMAX_ ? CMAX_ : s);
  }
  for (int i = 0; i < DF_; ++i) {
    float s = b2s[i];
    #pragma unroll
    for (int j = 0; j < DF_; ++j) s += u[j] * w2s[i * DF_ + j];
    v[i] = s < 0.f ? 0.f : (s > CMAX_ ? CMAX_ : s);
  }
  float o = b3s;
  #pragma unroll
  for (int j = 0; j < DF_; ++j) o += v[j] * w3s[j];
  out[row] = o;
}

extern "C" void kernel_launch(void* const* d_in, const int* in_sizes, int n_in,
                              void* d_out, int out_size, void* d_ws, size_t ws_size,
                              hipStream_t stream) {
  (void)in_sizes; (void)n_in; (void)out_size; (void)ws_size;
  const float* x    = (const float*)d_in[0];
  const float* W1   = (const float*)d_in[1];
  const float* b1   = (const float*)d_in[2];
  const float* W2   = (const float*)d_in[3];
  const float* b2   = (const float*)d_in[4];
  const float* W3   = (const float*)d_in[5];
  const float* b3   = (const float*)d_in[6];
  const float* W4   = (const float*)d_in[7];
  const float* b4   = (const float*)d_in[8];
  const float* fc1w = (const float*)d_in[9];
  const float* fc1b = (const float*)d_in[10];
  const float* fc2w = (const float*)d_in[11];
  const float* fc2b = (const float*)d_in[12];
  const float* fc3w = (const float*)d_in[13];
  const float* fc3b = (const float*)d_in[14];

  char* ws = (char*)d_ws;
  float*    featpart = (float*)(ws + FEAT_OFF);
  _Float16* w1e = (_Float16*)(ws + W1E_OFF);
  _Float16* w2e = (_Float16*)(ws + W2E_OFF);
  _Float16* w3e = (_Float16*)(ws + W3E_OFF);
  _Float16* w4e = (_Float16*)(ws + W4E_OFF);

  prep_w1<<<(unsigned)((W1E_N + 255) / 256), 256, 0, stream>>>(W1, w1e);
  prep_w2<<<(unsigned)((W2E_N + 255) / 256), 256, 0, stream>>>(W2, w2e);
  prep_w3<<<(unsigned)((W3E_N + 255) / 256), 256, 0, stream>>>(W3, w3e);
  prep_w4<<<(unsigned)((W4E_N + 255) / 256), 256, 0, stream>>>(W4, w4e);

  const size_t ldsBytes = (size_t)(ROWS * 32 + ROWS * 512) * sizeof(_Float16); // 139264
  tower_kernel<<<dim3(B_ / ROWS, 5), 128, ldsBytes, stream>>>(
      x, b1, b2, b3, b4, w1e, w2e, w3e, w4e, featpart);

  head_kernel<<<B_ / 256, 256, 0, stream>>>(featpart, fc1w, fc1b, fc2w, fc2b,
                                            fc3w, fc3b, (float*)d_out);
}
